// MultiHeadAttention_22677427323128
// MI455X (gfx1250) — matrix operand order repaired and checked by reading, not compiled, not run on hardware
//
#include <hip/hip_runtime.h>
#include <hip/hip_bf16.h>

typedef __attribute__((ext_vector_type(16))) _Float16 v16h;
typedef __attribute__((ext_vector_type(8)))  _Float16 v8h;
typedef __attribute__((ext_vector_type(4)))  _Float16 v4h;
typedef __attribute__((ext_vector_type(8)))  float    v8f;
typedef __attribute__((ext_vector_type(4)))  float    v4f;

#define DIM   1024
#define NHEAD 16
#define HD    64
#define SEQ   2048
#define BATCH 2
#define MTOT  (BATCH*SEQ)   // 4096

#define WMMA_F32_F16(a,b,c) \
  __builtin_amdgcn_wmma_f32_16x16x32_f16(false,(a),false,(b),(short)0,(c),false,false)

// A-matrix 16x32 f16 fragment (ISA 7.12.2): lane l<16 holds row l, K chunks {0..7,16..23};
// lanes>=16 hold row l-16, K chunks {8..15,24..31}. base already includes mrow*ld + kbase.
static __device__ inline v16h load_a_frag(const _Float16* base, int ld, int lane) {
  const int l16 = lane & 15, hf = lane >> 4;
  const _Float16* p = base + l16 * ld + hf * 8;
  union { v16h v; v8h h[2]; } u;
  u.h[0] = *(const v8h*)(p);
  u.h[1] = *(const v8h*)(p + 16);
  return u.v;
}

// B-matrix 32x16 f16 fragment with B = W^T (row-major W, N x K):
// lane l holds W-row (ncol + l%16), K chunks {8*(l/16)..+7, 16+8*(l/16)..+7}.
static __device__ inline v16h load_b_frag(const _Float16* base, int ld, int lane) {
  const int l16 = lane & 15, hf = lane >> 4;
  const _Float16* p = base + l16 * ld + hf * 8;
  union { v16h v; v8h h[2]; } u;
  u.h[0] = *(const v8h*)(p);
  u.h[1] = *(const v8h*)(p + 16);
  return u.v;
}

// ---------------------------------------------------------------- f32 -> f16
__global__ void cvt_f32_f16(const float* __restrict__ in, _Float16* __restrict__ out, int n4) {
  int i = blockIdx.x * 256 + threadIdx.x;
  if (i < n4) {
    v4f a = ((const v4f*)in)[i];
    v4h o;
#pragma unroll
    for (int j = 0; j < 4; ++j) o[j] = (_Float16)a[j];
    ((v4h*)out)[i] = o;
  }
}

// ------------------------------------------------- GEMM: Y = A (MTOTxK) * W^T (+bias)
// Per-wave 32x64 tile (2 M-tiles x 4 N-tiles); K loop unrolled 2x with ping-pong
// fragment buffers (no register rotation -> no mov/v_nop overhead).
// mode 0: f16 out, head-split  [b,h,s,hd]
// mode 1: f16 out, head-split transposed [b,h,hd,s]  (for V)
// mode 2: f32 out, flat [m,n] + bias      (final projection)
__global__ __launch_bounds__(128) void gemm_xwT(const _Float16* __restrict__ A,
                                                const _Float16* __restrict__ W,
                                                const float* __restrict__ bias,
                                                void* __restrict__ out, int mode) {
  const int lane = threadIdx.x & 31, wid = threadIdx.x >> 5;
  const int l16 = lane & 15, hf = lane >> 4;
  const int mrow0 = blockIdx.y * 128 + wid * 32;  // 32 M-rows per wave
  const int ncol  = blockIdx.x * 64;              // 4 N-tiles per wave

  const _Float16* Ab = A + (size_t)mrow0 * DIM;
  const _Float16* Wb = W + (size_t)ncol  * DIM;

  v8f acc[2][4] = {};
  v16h a0[2], b0[4], a1[2], b1[4];
#pragma unroll
  for (int mt = 0; mt < 2; ++mt) a0[mt] = load_a_frag(Ab + (size_t)mt * 16 * DIM, DIM, lane);
#pragma unroll
  for (int nt = 0; nt < 4; ++nt) b0[nt] = load_b_frag(Wb + (size_t)nt * 16 * DIM, DIM, lane);

  for (int kb = 0; kb < DIM; kb += 64) {
    // stage buf1 at kb+32 (always in range: kb <= DIM-64)
#pragma unroll
    for (int mt = 0; mt < 2; ++mt) a1[mt] = load_a_frag(Ab + (size_t)mt * 16 * DIM + kb + 32, DIM, lane);
#pragma unroll
    for (int nt = 0; nt < 4; ++nt) b1[nt] = load_b_frag(Wb + (size_t)nt * 16 * DIM + kb + 32, DIM, lane);
    if (kb + 256 < DIM) {
      __builtin_prefetch(Ab + kb + 256 + (size_t)l16 * DIM, 0, 1);
      __builtin_prefetch(Wb + kb + 256 + (size_t)l16 * DIM, 0, 1);
    }
#pragma unroll
    for (int mt = 0; mt < 2; ++mt)
#pragma unroll
      for (int nt = 0; nt < 4; ++nt)
        acc[mt][nt] = WMMA_F32_F16(a0[mt], b0[nt], acc[mt][nt]);

    // stage buf0 at kb+64 (wrap on last iteration; redundant load, harmless)
    const int k2 = (kb + 64 < DIM) ? (kb + 64) : 0;
#pragma unroll
    for (int mt = 0; mt < 2; ++mt) a0[mt] = load_a_frag(Ab + (size_t)mt * 16 * DIM + k2, DIM, lane);
#pragma unroll
    for (int nt = 0; nt < 4; ++nt) b0[nt] = load_b_frag(Wb + (size_t)nt * 16 * DIM + k2, DIM, lane);
#pragma unroll
    for (int mt = 0; mt < 2; ++mt)
#pragma unroll
      for (int nt = 0; nt < 4; ++nt)
        acc[mt][nt] = WMMA_F32_F16(a1[mt], b1[nt], acc[mt][nt]);
  }

#pragma unroll
  for (int mt = 0; mt < 2; ++mt)
#pragma unroll
    for (int nt = 0; nt < 4; ++nt)
#pragma unroll
      for (int r = 0; r < 8; ++r) {
        const int m = mrow0 + mt * 16 + r + 8 * hf;  // C layout: M = r + 8*(lane/16)
        const int n = ncol + nt * 16 + l16;          //           N = lane%16
        float val = acc[mt][nt][r];
        if (mode == 2) {
          ((float*)out)[(size_t)m * DIM + n] = val + bias[n];
        } else {
          if (bias) val += bias[n];
          const int b = m >> 11, s = m & (SEQ - 1);
          const int h = n >> 6, hd = n & (HD - 1);
          _Float16* o = (_Float16*)out;
          if (mode == 0)
            o[(((size_t)(b * NHEAD + h) * SEQ) + s) * HD + hd] = (_Float16)val;
          else
            o[((size_t)(b * NHEAD + h) * HD + hd) * SEQ + s] = (_Float16)val;
        }
      }
}

// ------------------------------------------- flash attention: one 16-row q-tile / wave
// K fragments ping-pong across 32-wide chunks (2 chunks per loop iteration, no copies);
// V fragments issued at the top of each chunk so their latency hides behind the score
// WMMAs and the softmax VALU/transcendental work.
__global__ __launch_bounds__(128) void attn_kernel(const _Float16* __restrict__ qh,
                                                   const _Float16* __restrict__ kh,
                                                   const _Float16* __restrict__ vT,
                                                   _Float16* __restrict__ aout) {
  __shared__ __align__(32) _Float16 pbuf[4][16 * 32];  // per-wave P tile (16x32 f16)
  const int lane = threadIdx.x & 31, wid = threadIdx.x >> 5;
  const int l16 = lane & 15, hf = lane >> 4;
  const int bh = blockIdx.x;                    // b*16 + h
  const int bb = bh >> 4, h = bh & 15;
  const int qbase = (blockIdx.y * 4 + wid) * 16;

  const _Float16* Q = qh + (size_t)bh * SEQ * HD;  // [s][hd]
  const _Float16* K = kh + (size_t)bh * SEQ * HD;  // [s][hd]
  const _Float16* V = vT + (size_t)bh * HD * SEQ;  // [hd][s]

  const v16h aq0 = load_a_frag(Q + (size_t)qbase * HD + 0,  HD, lane);
  const v16h aq1 = load_a_frag(Q + (size_t)qbase * HD + 32, HD, lane);

  v8f o[4] = {};
  float mr[8], lr[8];
#pragma unroll
  for (int r = 0; r < 8; ++r) { mr[r] = -1e30f; lr[r] = 0.f; }
  const float rs = 0.125f;                       // 1/sqrt(64)
  _Float16* pl = &pbuf[wid][0];

  // K fragment buffers: [j*2+c] covers N-tile j (kk rows), d-chunk c
  v16h kA[4], kB[4];
#pragma unroll
  for (int j = 0; j < 2; ++j)
#pragma unroll
    for (int c = 0; c < 2; ++c)
      kA[j * 2 + c] = load_b_frag(K + (size_t)(j * 16) * HD + c * 32, HD, lane);

  // one 32-wide chunk: scores with kcur, preload knxt at kpre, softmax, P x V
  auto chunk = [&](int kb, v16h* kcur, v16h* knxt, int kpre) {
    // V fragments for this chunk — issue first, consumed last
    v16h vf[4];
#pragma unroll
    for (int t = 0; t < 4; ++t)
      vf[t] = load_b_frag(V + (size_t)(t * 16) * SEQ + kb, SEQ, lane);

    // stage the other K buffer
#pragma unroll
    for (int j = 0; j < 2; ++j)
#pragma unroll
      for (int c = 0; c < 2; ++c)
        knxt[j * 2 + c] = load_b_frag(K + (size_t)(kpre + j * 16) * HD + c * 32, HD, lane);

    if (kb + 128 < SEQ) {
      __builtin_prefetch(K + (size_t)(kb + 128 + l16) * HD, 0, 1);
      __builtin_prefetch(V + (size_t)l16 * SEQ + kb + 128, 0, 1);
    }

    // scores tile 16 x 32
    v8f s0 = {}, s1 = {};
    s0 = WMMA_F32_F16(aq0, kcur[0], s0);
    s0 = WMMA_F32_F16(aq1, kcur[1], s0);
    s1 = WMMA_F32_F16(aq0, kcur[2], s1);
    s1 = WMMA_F32_F16(aq1, kcur[3], s1);

    // online softmax over this 32-wide chunk; row r maps to M = r + 8*hf
    float p0[8], p1[8], alpha[8];
#pragma unroll
    for (int r = 0; r < 8; ++r) { p0[r] = s0[r] * rs; p1[r] = s1[r] * rs; }
#pragma unroll
    for (int r = 0; r < 8; ++r) {
      float cm = fmaxf(p0[r], p1[r]);
      cm = fmaxf(cm, __shfl_xor(cm, 1, 16));
      cm = fmaxf(cm, __shfl_xor(cm, 2, 16));
      cm = fmaxf(cm, __shfl_xor(cm, 4, 16));
      cm = fmaxf(cm, __shfl_xor(cm, 8, 16));
      const float mn = fmaxf(mr[r], cm);
      alpha[r] = __expf(mr[r] - mn);
      mr[r] = mn;
      p0[r] = __expf(p0[r] - mn);
      p1[r] = __expf(p1[r] - mn);
      float cs = p0[r] + p1[r];
      cs += __shfl_xor(cs, 1, 16);
      cs += __shfl_xor(cs, 2, 16);
      cs += __shfl_xor(cs, 4, 16);
      cs += __shfl_xor(cs, 8, 16);
      lr[r] = lr[r] * alpha[r] + cs;
    }
#pragma unroll
    for (int t = 0; t < 4; ++t)
#pragma unroll
      for (int r = 0; r < 8; ++r) o[t][r] *= alpha[r];

    // C-layout -> A-layout via per-wave LDS tile (compiler orders the DS RAW)
#pragma unroll
    for (int r = 0; r < 8; ++r) {
      pl[(r + 8 * hf) * 32 + l16]      = (_Float16)p0[r];
      pl[(r + 8 * hf) * 32 + 16 + l16] = (_Float16)p1[r];
    }
    v16h ap = load_a_frag(pl, 32, lane);

    // O += P (16x32) x V (32x64)
#pragma unroll
    for (int t = 0; t < 4; ++t) o[t] = WMMA_F32_F16(ap, vf[t], o[t]);
  };

  for (int kb = 0; kb < SEQ; kb += 64) {
    chunk(kb, kA, kB, kb + 32);                        // kb+32 <= SEQ-32, in range
    const int k2 = (kb + 64 < SEQ) ? (kb + 64) : 0;    // wrap on last iteration
    chunk(kb + 32, kB, kA, k2);
  }

#pragma unroll
  for (int t = 0; t < 4; ++t)
#pragma unroll
    for (int r = 0; r < 8; ++r) {
      const int m = bb * SEQ + qbase + r + 8 * hf;
      aout[(size_t)m * DIM + h * HD + t * 16 + l16] = (_Float16)(o[t][r] / lr[r]);
    }
}

// ----------------------------------------------------------------------------
extern "C" void kernel_launch(void* const* d_in, const int* in_sizes, int n_in,
                              void* d_out, int out_size, void* d_ws, size_t ws_size,
                              hipStream_t stream) {
  const float* x  = (const float*)d_in[0];
  const float* Wq = (const float*)d_in[1];
  const float* Wk = (const float*)d_in[2];
  const float* bk = (const float*)d_in[3];
  const float* Wv = (const float*)d_in[4];
  const float* Wo = (const float*)d_in[5];
  const float* bo = (const float*)d_in[6];

  char* ws = (char*)d_ws;
  const size_t MB = 1024 * 1024;
  _Float16* xh  = (_Float16*)(ws + 0 * MB);   // 4096x1024 f16 (8 MB)
  _Float16* Wqh = (_Float16*)(ws + 8 * MB);   // 2 MB each
  _Float16* Wkh = (_Float16*)(ws + 10 * MB);
  _Float16* Wvh = (_Float16*)(ws + 12 * MB);
  _Float16* Woh = (_Float16*)(ws + 14 * MB);
  _Float16* qh  = (_Float16*)(ws + 16 * MB);  // [b,h,s,hd] f16 (8 MB)
  _Float16* kh  = (_Float16*)(ws + 24 * MB);  // [b,h,s,hd]
  _Float16* vT  = (_Float16*)(ws + 32 * MB);  // [b,h,hd,s]
  _Float16* ah  = (_Float16*)(ws + 40 * MB);  // [m, dim] f16 attn output

  const int nx = MTOT * DIM, nw = DIM * DIM;
  cvt_f32_f16<<<(nx / 4 + 255) / 256, 256, 0, stream>>>(x,  xh,  nx / 4);
  cvt_f32_f16<<<(nw / 4 + 255) / 256, 256, 0, stream>>>(Wq, Wqh, nw / 4);
  cvt_f32_f16<<<(nw / 4 + 255) / 256, 256, 0, stream>>>(Wk, Wkh, nw / 4);
  cvt_f32_f16<<<(nw / 4 + 255) / 256, 256, 0, stream>>>(Wv, Wvh, nw / 4);
  cvt_f32_f16<<<(nw / 4 + 255) / 256, 256, 0, stream>>>(Wo, Woh, nw / 4);

  dim3 gg(DIM / 64, MTOT / 128);                // (16, 32)
  gemm_xwT<<<gg, 128, 0, stream>>>(xh, Wqh, nullptr, qh, 0);
  gemm_xwT<<<gg, 128, 0, stream>>>(xh, Wkh, bk,      kh, 0);
  gemm_xwT<<<gg, 128, 0, stream>>>(xh, Wvh, nullptr, vT, 1);

  dim3 ga(BATCH * NHEAD, SEQ / 64);             // (32, 32), 4 q-tiles per block
  attn_kernel<<<ga, 128, 0, stream>>>(qh, kh, vT, ah);

  gemm_xwT<<<gg, 128, 0, stream>>>(ah, Woh, bo, d_out, 2);
}
